// HiGNN_17059610100234
// MI455X (gfx1250) — matrix operand order repaired and checked by reading, not compiled, not run on hardware
//
#include <hip/hip_runtime.h>

// ---------------------------------------------------------------------------
// GraphCast-style hierarchical GNN forward for MI455X (gfx1250, wave32, WMMA).
//
//   msg1(concat[src,dst,ef]) = src@Wa + dst@Wb + ef@Wc + b1  (row slices of
//   msg1.W).  Node-level GEMMs run as dense bf16 WMMA GEMMs; the per-edge
//   msg2 GEMM is fused with relu-edge-assembly (A-tile gathered on the fly)
//   and with segment_sum (global_atomic_add_f32 epilogue into agg[dst]).
//
//   Weights are converted once per launch to bf16, pre-transposed [M][Kpad]
//   (zero-padded K): a B fragment is two 16B runs per lane -> two
//   global_load_b128 from L2-resident weights.  The dense GEMM is LDS-free
//   and barrier-free; all four B fragments of a K-step are prefetched in one
//   clause before the 8 back-to-back wmmas; FAST_A / TAIL are template
//   parameters so no uniform branches or dead paths remain in the loop.
//
//   GEMM core: 256 threads (8 waves), block tile 128x128, wave tile 32x64,
//   K-step 32 -> 8 v_wmma_f32_16x16x32_bf16 per wave per K-step.
// ---------------------------------------------------------------------------

typedef __attribute__((ext_vector_type(16))) __bf16 v16bf;
typedef __attribute__((ext_vector_type(8)))  __bf16 v8bf;
typedef __attribute__((ext_vector_type(8)))  float  v8f;

#define HD 128

__device__ __forceinline__ v8bf pack8(float a0, float a1, float a2, float a3,
                                      float a4, float a5, float a6, float a7)
{
    v8bf p;
    p[0] = (__bf16)a0; p[1] = (__bf16)a1; p[2] = (__bf16)a2; p[3] = (__bf16)a3;
    p[4] = (__bf16)a4; p[5] = (__bf16)a5; p[6] = (__bf16)a6; p[7] = (__bf16)a7;
    return p;
}

// A fragment (ISA 16-bit A 16x32 layout) = two contiguous 8-float runs of one
// row: k = kb..kb+7 (elements 0..7) and k = 16+kb..16+kb+7 (elements 8..15),
// kb = 0 (lanes 0-15) or 8 (lanes 16-31).  rowp already includes kb.
template <bool FAST>
__device__ __forceinline__ v16bf make_afrag(const float* __restrict__ rowp)
{
    float f[16];
    if (FAST) {
        float4 a = *reinterpret_cast<const float4*>(rowp);
        float4 b = *reinterpret_cast<const float4*>(rowp + 4);
        float4 c = *reinterpret_cast<const float4*>(rowp + 16);
        float4 d = *reinterpret_cast<const float4*>(rowp + 20);
        f[0]=a.x; f[1]=a.y; f[2]=a.z;  f[3]=a.w;  f[4]=b.x;  f[5]=b.y;  f[6]=b.z;  f[7]=b.w;
        f[8]=c.x; f[9]=c.y; f[10]=c.z; f[11]=c.w; f[12]=d.x; f[13]=d.y; f[14]=d.z; f[15]=d.w;
    } else {
        #pragma unroll
        for (int j = 0; j < 8; ++j) { f[j] = rowp[j]; f[8 + j] = rowp[16 + j]; }
    }
    v16bf r;
    #pragma unroll
    for (int j = 0; j < 16; ++j) r[j] = (__bf16)f[j];
    return r;
}

// A fragment from LDS row (already bf16, contiguous 32-k row).
__device__ __forceinline__ v16bf load_afrag_lds(const __bf16* rowp, int lane)
{
    const int kb = (lane >> 4) ? 8 : 0;
    v16bf a;
    #pragma unroll
    for (int v = 0; v < 4; ++v) {
        a[2 * v]     = rowp[kb + 2 * v];
        a[2 * v + 1] = rowp[kb + 2 * v + 1];
        a[2 * v + 8] = rowp[16 + kb + 2 * v];
        a[2 * v + 9] = rowp[16 + kb + 2 * v + 1];
    }
    return a;
}

// B fragment: two 16B runs per lane (k = kb..kb+7, 16+kb..16+kb+7) from pre-transposed [M][Kpad] bf16.
__device__ __forceinline__ v16bf load_bfrag_g(const __bf16* colp, int kc)
{
    v8bf lo = *reinterpret_cast<const v8bf*>(colp + kc);
    v8bf hi = *reinterpret_cast<const v8bf*>(colp + kc + 16);
    v16bf b;
    #pragma unroll
    for (int i = 0; i < 8; ++i) { b[i] = lo[i]; b[8 + i] = hi[i]; }
    return b;
}

// ---------------------------------------------------------------------------
// Dense GEMM: out[N,M] = A[N,K] @ W[K,M] (+bias) (+addin); W pre-transposed
// bf16 [M][ldk].  LDS-free, barrier-free.
// ---------------------------------------------------------------------------
template <bool FAST_A, bool TAIL>
__global__ __launch_bounds__(256)
void gemm_bf16_wmma(const float* __restrict__ A, int lda,
                    const __bf16* __restrict__ WT, int ldk,
                    const float* __restrict__ bias,
                    const float* __restrict__ addin,
                    float* __restrict__ out, int ldo,
                    int N, int K, int M)
{
    const int row0 = blockIdx.x * 128;
    const int col0 = blockIdx.y * 128;
    const int tid  = threadIdx.x;
    const int lane = tid & 31;
    const int wv   = tid >> 5;
    const int wr   = (wv & 3) * 32;    // wave row offset (2 row-groups of 16)
    const int wc   = (wv >> 2) * 64;   // wave col offset (4 col tiles of 16)
    const int l15  = lane & 15;
    const int kbA  = (lane >> 4) ? 8 : 0;
    const int kbB  = (lane >> 4) ? 8 : 0;

    const int r0 = min(row0 + wr + l15,      N - 1);
    const int r1 = min(row0 + wr + 16 + l15, N - 1);
    const float* arow0 = A + (size_t)r0 * lda + kbA;
    const float* arow1 = A + (size_t)r1 * lda + kbA;

    const __bf16* bp[4];
    #pragma unroll
    for (int t = 0; t < 4; ++t) {
        int gc = min(col0 + wc + t * 16 + l15, M - 1);
        bp[t] = WT + (size_t)gc * ldk + kbB;
    }

    v8f acc[2][4] = {};
    const int kfull = K & ~31;

    for (int kc = 0; kc < kfull; kc += 32) {
        // Prefetch all four B fragments (one clause, one wait) ...
        v16bf bfr[4];
        #pragma unroll
        for (int t = 0; t < 4; ++t) bfr[t] = load_bfrag_g(bp[t], kc);
        // ... and both A fragments.
        v16bf af0 = make_afrag<FAST_A>(arow0 + kc);
        v16bf af1 = make_afrag<FAST_A>(arow1 + kc);
        // 8 back-to-back wmmas.
        #pragma unroll
        for (int t = 0; t < 4; ++t) {
            acc[0][t] = __builtin_amdgcn_wmma_f32_16x16x32_bf16(
                false, af0, false, bfr[t], (short)0, acc[0][t], false, false);
            acc[1][t] = __builtin_amdgcn_wmma_f32_16x16x32_bf16(
                false, af1, false, bfr[t], (short)0, acc[1][t], false, false);
        }
    }
    if (TAIL && kfull < K) {            // single K-tail chunk (encoder only)
        float f0[16], f1[16];
        #pragma unroll
        for (int j = 0; j < 8; ++j) {
            int gkl = kfull + kbA + j;        // low run
            int gkh = kfull + 16 + kbA + j;   // high run
            float vl0 = A[(size_t)r0 * lda + min(gkl, K - 1)];
            float vh0 = A[(size_t)r0 * lda + min(gkh, K - 1)];
            float vl1 = A[(size_t)r1 * lda + min(gkl, K - 1)];
            float vh1 = A[(size_t)r1 * lda + min(gkh, K - 1)];
            f0[j] = (gkl < K) ? vl0 : 0.f;  f0[8 + j] = (gkh < K) ? vh0 : 0.f;
            f1[j] = (gkl < K) ? vl1 : 0.f;  f1[8 + j] = (gkh < K) ? vh1 : 0.f;
        }
        v16bf af0, af1;
        #pragma unroll
        for (int j = 0; j < 16; ++j) { af0[j] = (__bf16)f0[j]; af1[j] = (__bf16)f1[j]; }
        v16bf bfr[4];
        #pragma unroll
        for (int t = 0; t < 4; ++t) bfr[t] = load_bfrag_g(bp[t], kfull);
        #pragma unroll
        for (int t = 0; t < 4; ++t) {   // WT zero-padded beyond K
            acc[0][t] = __builtin_amdgcn_wmma_f32_16x16x32_bf16(
                false, af0, false, bfr[t], (short)0, acc[0][t], false, false);
            acc[1][t] = __builtin_amdgcn_wmma_f32_16x16x32_bf16(
                false, af1, false, bfr[t], (short)0, acc[1][t], false, false);
        }
    }

    // C/D layout: VGPR r, lanes0-15: M=r; lanes16-31: M=r+8; N=lane&15
    const int rsel = (lane >> 4) ? 8 : 0;
    float bv[4];
    #pragma unroll
    for (int t = 0; t < 4; ++t) {
        int gc = col0 + wc + t * 16 + l15;
        bv[t] = bias ? bias[min(gc, M - 1)] : 0.f;
    }
    #pragma unroll
    for (int g = 0; g < 2; ++g) {
        #pragma unroll
        for (int r = 0; r < 8; ++r) {
            int gr = row0 + wr + g * 16 + rsel + r;
            if (gr >= N) continue;
            #pragma unroll
            for (int t = 0; t < 4; ++t) {
                int gc = col0 + wc + t * 16 + l15;
                if (gc >= M) continue;
                float v = acc[g][t][r] + bv[t];
                if (addin) v += addin[(size_t)gr * ldo + gc];
                out[(size_t)gr * ldo + gc] = v;
            }
        }
    }
}

// ---------------------------------------------------------------------------
// Fused edge kernel: h = relu(Xa[src] + Xb[dst] + ef@Wc + b1) built on the
// fly as the WMMA A-tile (ping-pong LDS, one barrier per K-step), times W2T
// (pre-transposed bf16) + b2, scatter-added into agg[dst].  K = M = 128.
// ---------------------------------------------------------------------------
__global__ __launch_bounds__(256)
void msg_scatter_wmma(const float* __restrict__ Xa,
                      const float* __restrict__ Xb,
                      const int*  __restrict__ ei_src,
                      const int*  __restrict__ ei_dst,
                      const float* __restrict__ ef,    // [E,4]
                      const float* __restrict__ Wc,    // [4,128] fp32
                      const float* __restrict__ b1,    // [128]
                      const __bf16* __restrict__ W2T,  // [128][128] bf16
                      const float* __restrict__ b2,    // [128]
                      float* __restrict__ agg,         // [n_dst,128]
                      int E)
{
    __shared__ __align__(16) __bf16 sA[2][64][40];

    const int row0 = blockIdx.x * 64;   // edge base
    const int tid  = threadIdx.x;
    const int lane = tid & 31;
    const int wv   = tid >> 5;
    const int wr   = (wv & 3) * 16;
    const int wc   = (wv >> 2) * 64;
    const int l15  = lane & 15;

    const int ar  = tid >> 2;           // edge row in tile
    const int akq = (tid & 3) * 8;      // feature sub-offset

    const int e0  = min(row0 + ar, E - 1);
    const int sI  = ei_src[e0];
    const int dI  = ei_dst[e0];
    const float4 efv = *reinterpret_cast<const float4*>(ef + (size_t)e0 * 4);
    const float* xaP = Xa + (size_t)sI * HD + akq;
    const float* xbP = Xb + (size_t)dI * HD + akq;

    const int kb2 = (lane >> 4) ? 8 : 0;
    const __bf16* bp[4];
    #pragma unroll
    for (int t = 0; t < 4; ++t)
        bp[t] = W2T + (size_t)(wc + t * 16 + l15) * HD + kb2;

    v8f acc[4] = {};

    int ib = 0;
    for (int kc = 0; kc < HD; kc += 32, ib ^= 1) {
        {   // fused A-tile: 8 features of one edge per thread
            const int gk = kc + akq;
            const float4* pa  = reinterpret_cast<const float4*>(xaP + kc);
            const float4* pb  = reinterpret_cast<const float4*>(xbP + kc);
            const float4* pc0 = reinterpret_cast<const float4*>(Wc + 0 * HD + gk);
            const float4* pc1 = reinterpret_cast<const float4*>(Wc + 1 * HD + gk);
            const float4* pc2 = reinterpret_cast<const float4*>(Wc + 2 * HD + gk);
            const float4* pc3 = reinterpret_cast<const float4*>(Wc + 3 * HD + gk);
            const float4* pbi = reinterpret_cast<const float4*>(b1 + gk);
            float v[8];
            #pragma unroll
            for (int h = 0; h < 2; ++h) {
                float4 xa = pa[h], xb = pb[h], bb = pbi[h];
                float4 c0 = pc0[h], c1 = pc1[h], c2 = pc2[h], c3 = pc3[h];
                v[4*h+0] = fmaxf(xa.x + xb.x + bb.x + efv.x*c0.x + efv.y*c1.x + efv.z*c2.x + efv.w*c3.x, 0.f);
                v[4*h+1] = fmaxf(xa.y + xb.y + bb.y + efv.x*c0.y + efv.y*c1.y + efv.z*c2.y + efv.w*c3.y, 0.f);
                v[4*h+2] = fmaxf(xa.z + xb.z + bb.z + efv.x*c0.z + efv.y*c1.z + efv.z*c2.z + efv.w*c3.z, 0.f);
                v[4*h+3] = fmaxf(xa.w + xb.w + bb.w + efv.x*c0.w + efv.y*c1.w + efv.z*c2.w + efv.w*c3.w, 0.f);
            }
            *reinterpret_cast<v8bf*>(&sA[ib][ar][akq]) =
                pack8(v[0], v[1], v[2], v[3], v[4], v[5], v[6], v[7]);
        }
        __syncthreads();    // ping-pong: single barrier per K-step
        v16bf bfr[4];
        #pragma unroll
        for (int t = 0; t < 4; ++t) bfr[t] = load_bfrag_g(bp[t], kc);
        v16bf af = load_afrag_lds(&sA[ib][wr + l15][0], lane);
        #pragma unroll
        for (int t = 0; t < 4; ++t) {
            acc[t] = __builtin_amdgcn_wmma_f32_16x16x32_bf16(
                false, af, false, bfr[t], (short)0, acc[t], false, false);
        }
    }

    // Fused segment_sum: global_atomic_add_f32 into agg[dst]
    const int rsel = (lane >> 4) ? 8 : 0;
    float b2v[4];
    #pragma unroll
    for (int t = 0; t < 4; ++t) b2v[t] = b2[wc + t * 16 + l15];
    #pragma unroll
    for (int r = 0; r < 8; ++r) {
        int e = row0 + wr + rsel + r;
        if (e >= E) continue;
        int d = ei_dst[e];
        #pragma unroll
        for (int t = 0; t < 4; ++t) {
            int gc = wc + t * 16 + l15;
            atomicAdd(&agg[(size_t)d * HD + gc], acc[t][r] + b2v[t]);
        }
    }
}

// ---------------------------------------------------------------------------
// Small kernels
// ---------------------------------------------------------------------------
__global__ void cvt_wT_kernel(const float* __restrict__ W,
                              __bf16* __restrict__ out,
                              int K, int M, int Kpad)
{
    int i = blockIdx.x * blockDim.x + threadIdx.x;
    if (i >= M * Kpad) return;
    int m = i / Kpad, k = i - m * Kpad;
    out[i] = (k < K) ? (__bf16)W[(size_t)k * M + m] : (__bf16)0.f;
}

__global__ void mesh_enc_kernel(const float* __restrict__ mf,   // [n,2]
                                const float* __restrict__ W,    // [2,128]
                                const float* __restrict__ b,    // [128]
                                float* __restrict__ out, int n)
{
    int i = blockIdx.x * blockDim.x + threadIdx.x;
    if (i >= n * HD) return;
    int node = i >> 7, f = i & (HD - 1);
    out[i] = mf[node * 2 + 0] * W[f] + mf[node * 2 + 1] * W[HD + f] + b[f];
}

__global__ void zero_kernel(float* __restrict__ p, int n)
{
    int i = blockIdx.x * blockDim.x + threadIdx.x;
    if (i < n) p[i] = 0.f;
}

// ---------------------------------------------------------------------------
// Host orchestration
// ---------------------------------------------------------------------------
extern "C" void kernel_launch(void* const* d_in, const int* in_sizes, int n_in,
                              void* d_out, int out_size, void* d_ws, size_t ws_size,
                              hipStream_t stream)
{
    (void)in_sizes; (void)n_in; (void)out_size; (void)ws_size;

    const int NG = 65536;
    const int NM[3] = {642, 2562, 10242};
    const int EG2M = 262144, EM2G = 262144;
    const int EM2M[3] = {3840, 15360, 61440};
    const int EUP[2]  = {10240, 40960};
    const int EDN[2]  = {10240, 40960};

    // ----- graph inputs (setup_inputs dict order) -----
    const float* grid_features = (const float*)d_in[0];
    const int*   g2m_ei = (const int*)d_in[1];
    const float* g2m_ef = (const float*)d_in[2];
    const int*   m2g_ei = (const int*)d_in[3];
    const float* m2g_ef = (const float*)d_in[4];
    const int*   m2m_ei[3]; const float* m2m_ef[3]; const float* mesh_f[3];
    for (int i = 0; i < 3; ++i) {
        m2m_ei[i] = (const int*)d_in[5 + 3 * i];
        m2m_ef[i] = (const float*)d_in[6 + 3 * i];
        mesh_f[i] = (const float*)d_in[7 + 3 * i];
    }
    const int* up_ei[2]; const float* up_ef[2];
    const int* dn_ei[2]; const float* dn_ef[2];
    up_ei[0] = (const int*)d_in[14]; up_ef[0] = (const float*)d_in[15];
    dn_ei[0] = (const int*)d_in[16]; dn_ef[0] = (const float*)d_in[17];
    up_ei[1] = (const int*)d_in[18]; up_ef[1] = (const float*)d_in[19];
    dn_ei[1] = (const int*)d_in[20]; dn_ef[1] = (const float*)d_in[21];

    // ----- params: jax pytree flatten order (dict keys sorted) -----
    auto L = [&](int i) { return (const float*)d_in[i]; };
    struct Mpl { const float *m1W, *m1b, *m2W, *m2b, *uW, *ub; };
    auto mplAt = [&](int b) {
        return Mpl{L(b), L(b + 1), L(b + 2), L(b + 3), L(b + 4), L(b + 5)};
    };
    Mpl p_down[2] = { mplAt(22), mplAt(28) };
    Mpl p_g2m     = mplAt(34);
    const float* decW = L(40); const float* decB = L(41);
    const float* encW = L(42); const float* encB = L(43);
    Mpl p_m2g     = mplAt(44);
    const float* meW[3] = {L(50), L(52), L(54)};
    const float* meB[3] = {L(51), L(53), L(55)};
    Mpl p_same[3] = { mplAt(56), mplAt(62), mplAt(68) };
    Mpl p_up[2]   = { mplAt(74), mplAt(80) };

    // ----- workspace: fp32 activations, then bf16 transposed weights -----
    float* w = (float*)d_ws;
    size_t off = 0;
    auto alloc = [&](size_t n) { float* p = w + off; off += n; return p; };
    float* grid_a = alloc((size_t)NG * HD);
    float* grid_b = alloc((size_t)NG * HD);
    float* Xa     = alloc((size_t)NG * HD);
    float* Xb     = alloc((size_t)NG * HD);
    float* agg    = alloc((size_t)NG * HD);
    float* mesh_buf[3][2];
    for (int i = 0; i < 3; ++i) {
        size_t pad = (size_t)((NM[i] + 63) / 64) * 64 * HD;
        mesh_buf[i][0] = alloc(pad);
        mesh_buf[i][1] = alloc(pad);
    }
    int cur[3] = {0, 0, 0};

    __bf16* bw = (__bf16*)(w + off);
    size_t boff = 0;
    auto cvt = [&](const float* W_, int K, int M) {
        int Kpad = (K + 31) & ~31;
        __bf16* p = bw + boff;
        boff += (size_t)M * Kpad;
        int n = M * Kpad;
        cvt_wT_kernel<<<(n + 255) / 256, 256, 0, stream>>>(W_, p, K, M, Kpad);
        return p;
    };

    struct MplT { const __bf16 *wTa, *wTb, *wT2, *wTua, *wTub; };
    auto cvtMpl = [&](const Mpl& p) {
        MplT t;
        t.wTa  = cvt(p.m1W,            HD, HD);   // msg1 rows   0..127
        t.wTb  = cvt(p.m1W + 128 * HD, HD, HD);   // msg1 rows 128..255
        t.wT2  = cvt(p.m2W,            HD, HD);
        t.wTua = cvt(p.uW,             HD, HD);   // upd  rows   0..127
        t.wTub = cvt(p.uW + 128 * HD,  HD, HD);   // upd  rows 128..255
        return t;
    };
    MplT t_down[2] = { cvtMpl(p_down[0]), cvtMpl(p_down[1]) };
    MplT t_g2m     = cvtMpl(p_g2m);
    MplT t_m2g     = cvtMpl(p_m2g);
    MplT t_same[3] = { cvtMpl(p_same[0]), cvtMpl(p_same[1]), cvtMpl(p_same[2]) };
    MplT t_up[2]   = { cvtMpl(p_up[0]), cvtMpl(p_up[1]) };
    const __bf16* encWT = cvt(encW, 78, HD);      // Kpad = 96
    const __bf16* decWT = cvt(decW, HD, 78);

    // ----- launch helpers -----
    auto gemm = [&](const float* A, const __bf16* WT, const float* bias,
                    const float* addin, float* out, int N, int K, int M) {
        int Kpad = (K + 31) & ~31;
        dim3 g((N + 127) / 128, (M + 127) / 128);
        bool fast = ((K & 3) == 0);        // lda == K
        bool tail = ((K & 31) != 0);
        if (fast && !tail)
            gemm_bf16_wmma<true, false><<<g, 256, 0, stream>>>(
                A, K, WT, Kpad, bias, addin, out, M, N, K, M);
        else if (fast && tail)
            gemm_bf16_wmma<true, true><<<g, 256, 0, stream>>>(
                A, K, WT, Kpad, bias, addin, out, M, N, K, M);
        else if (!fast && tail)
            gemm_bf16_wmma<false, true><<<g, 256, 0, stream>>>(
                A, K, WT, Kpad, bias, addin, out, M, N, K, M);
        else
            gemm_bf16_wmma<false, false><<<g, 256, 0, stream>>>(
                A, K, WT, Kpad, bias, addin, out, M, N, K, M);
    };
    auto mpl = [&](const Mpl& p, const MplT& t, const float* src, int n_src,
                   const float* dst, int n_dst, const int* ei, int E,
                   const float* ef, float* outbuf) {
        gemm(src, t.wTa, nullptr, nullptr, Xa, n_src, HD, HD);
        gemm(dst, t.wTb, nullptr, nullptr, Xb, n_dst, HD, HD);
        int nz = n_dst * HD;
        zero_kernel<<<(nz + 255) / 256, 256, 0, stream>>>(agg, nz);
        msg_scatter_wmma<<<(E + 63) / 64, 256, 0, stream>>>(
            Xa, Xb, ei, ei + E, ef, p.m1W + 256 * HD, p.m1b,
            t.wT2, p.m2b, agg, E);
        gemm(dst, t.wTua, p.ub,    dst,    outbuf, n_dst, HD, HD);
        gemm(agg, t.wTub, nullptr, outbuf, outbuf, n_dst, HD, HD);
    };
    auto mcur  = [&](int i) { return mesh_buf[i][cur[i]]; };
    auto mnext = [&](int i) { return mesh_buf[i][cur[i] ^ 1]; };
    auto flip  = [&](int i) { cur[i] ^= 1; };

    // ----- forward -----
    gemm(grid_features, encWT, encB, nullptr, grid_a, NG, 78, HD);
    for (int i = 0; i < 3; ++i) {
        int nt = NM[i] * HD;
        mesh_enc_kernel<<<(nt + 255) / 256, 256, 0, stream>>>(
            mesh_f[i], meW[i], meB[i], mesh_buf[i][0], NM[i]);
    }

    mpl(p_g2m, t_g2m, grid_a, NG, mcur(2), NM[2], g2m_ei, EG2M, g2m_ef, mnext(2)); flip(2);
    mpl(p_same[2], t_same[2], mcur(2), NM[2], mcur(2), NM[2], m2m_ei[2], EM2M[2], m2m_ef[2], mnext(2)); flip(2);
    mpl(p_up[1],   t_up[1],   mcur(2), NM[2], mcur(1), NM[1], up_ei[1],  EUP[1],  up_ef[1],  mnext(1)); flip(1);
    mpl(p_same[1], t_same[1], mcur(1), NM[1], mcur(1), NM[1], m2m_ei[1], EM2M[1], m2m_ef[1], mnext(1)); flip(1);
    mpl(p_up[0],   t_up[0],   mcur(1), NM[1], mcur(0), NM[0], up_ei[0],  EUP[0],  up_ef[0],  mnext(0)); flip(0);
    mpl(p_same[0], t_same[0], mcur(0), NM[0], mcur(0), NM[0], m2m_ei[0], EM2M[0], m2m_ef[0], mnext(0)); flip(0);
    mpl(p_down[0], t_down[0], mcur(0), NM[0], mcur(1), NM[1], dn_ei[0],  EDN[0],  dn_ef[0],  mnext(1)); flip(1);
    mpl(p_same[1], t_same[1], mcur(1), NM[1], mcur(1), NM[1], m2m_ei[1], EM2M[1], m2m_ef[1], mnext(1)); flip(1);
    mpl(p_down[1], t_down[1], mcur(1), NM[1], mcur(2), NM[2], dn_ei[1],  EDN[1],  dn_ef[1],  mnext(2)); flip(2);
    mpl(p_same[2], t_same[2], mcur(2), NM[2], mcur(2), NM[2], m2m_ei[2], EM2M[2], m2m_ef[2], mnext(2)); flip(2);
    mpl(p_m2g, t_m2g, mcur(2), NM[2], grid_a, NG, m2g_ei, EM2G, m2g_ef, grid_b);
    gemm(grid_b, decWT, decB, nullptr, (float*)d_out, NG, HD, 78);
}
